// GumbelRNNGenerator_35467839931073
// MI455X (gfx1250) — matrix operand order repaired and checked by reading, not compiled, not run on hardware
//
#include <hip/hip_runtime.h>
#include <stdint.h>

// ---------------- constants ----------------
#define B      256
#define HID    512
#define NOISE  128
#define VOCAB  5000
#define VPAD   5008          // padded logits/h2o width (multiple of 16)
#define G3     1536          // 3*HID
#define SOS    4999
#define EPSN   1e-5f

typedef __attribute__((ext_vector_type(16))) __bf16 v16bf;
typedef __attribute__((ext_vector_type(8)))  float  v8f;

union FragBF {
    v16bf v;
    uint4 u[2];
};

// ---------------- helpers ----------------
__device__ __forceinline__ unsigned short f32_to_bf16(float f) {
    unsigned u = __float_as_uint(f);
    unsigned r = u + 0x7FFFu + ((u >> 16) & 1u);   // round-to-nearest-even
    return (unsigned short)(r >> 16);
}

__device__ __forceinline__ float lrelu(float x) { return x >= 0.f ? x : 0.2f * x; }
__device__ __forceinline__ float sigmoidf_(float x) { return 1.f / (1.f + expf(-x)); }

__device__ __forceinline__ float blockSum256(float v) {
    __shared__ float s[256];
    int t = threadIdx.x;
    s[t] = v;
    __syncthreads();
#pragma unroll
    for (int off = 128; off > 0; off >>= 1) {
        if (t < off) s[t] += s[t + off];
        __syncthreads();
    }
    float r = s[0];
    __syncthreads();
    return r;
}

__device__ __forceinline__ void stats256(float x, float& mu, float& va) {
    mu = blockSum256(x) * (1.f / 256.f);
    float d = x - mu;
    va = blockSum256(d * d) * (1.f / 256.f);
}

// ---------------- f32 -> bf16 convert (with zero pad) ----------------
__global__ void k_cvt_bf16(const float* __restrict__ src, unsigned short* __restrict__ dst,
                           int n, int ntot) {
    int i = blockIdx.x * blockDim.x + threadIdx.x;
    if (i < n)        dst[i] = f32_to_bf16(src[i]);
    else if (i < ntot) dst[i] = 0;
}

// ---------------- bf16 WMMA GEMM: C[M,N] = A[M,K] * W[N,K]^T + bias ----------------
// Block: 256 threads = 8 waves; wave tile 16M x 64N; block tile 32M x 256N.
// A panel (32 x K bf16, contiguous) staged in LDS via async global->LDS (ASYNCcnt),
// consumed with ds_load_b128. B fragments from global (L2-resident weights).
// K loop is software-pipelined: next k-step's fragments are loaded before the
// current k-step's 4 WMMAs so load latency overlaps matrix issue.
__global__ __launch_bounds__(256)
void k_gemm_bf16(const unsigned short* __restrict__ A,   // [M,K] row-major bf16
                 const unsigned short* __restrict__ W,   // [Nw,K] row-major bf16
                 const float* __restrict__ bias,         // [Nreal]
                 float* __restrict__ C,                  // [M,N] row-major f32
                 int N, int Nw, int Nreal, int K) {
    extern __shared__ unsigned short smemA[];            // 32*K bf16

    const int wave = threadIdx.x >> 5;
    const int lane = threadIdx.x & 31;
    const int wm = wave >> 2;           // 0..1
    const int wn = wave & 3;            // 0..3
    const int m0 = blockIdx.y * 32 + wm * 16;
    const int nb = blockIdx.x * 256 + wn * 64;

    const int lrow  = lane & 15;
    const int hi    = lane >> 4;        // 0 or 1
    const int kaoff = hi << 3;          // A: +0 / +8
    const int kboff = hi << 3;          // B: +0 / +8

    // ---- async stage A panel: rows [blockIdx.y*32, +32) x K, contiguous ----
    {
        const unsigned short* Ablk = A + (size_t)blockIdx.y * 32 * K;
        const int nseg = (32 * K) >> 3;                 // 16B segments
        uint32_t lds_base = (uint32_t)(uintptr_t)(&smemA[0]);
        for (int s = threadIdx.x; s < nseg; s += 256) {
            uint32_t lds_v = lds_base + (uint32_t)(s << 4);
            const void* gp = (const void*)((const char*)Ablk + ((size_t)s << 4));
            asm volatile("global_load_async_to_lds_b128 %0, %1, off"
                         :: "v"(lds_v), "v"(gp) : "memory");
        }
        asm volatile("s_wait_asynccnt 0x0" ::: "memory");
    }
    __syncthreads();

    // weight row pointers (clamped -> branch-free loads; invalid cols never stored)
    int colv[4];
    const unsigned short* wrow[4];
#pragma unroll
    for (int i = 0; i < 4; ++i) {
        colv[i] = nb + i * 16 + lrow;
        int colL = colv[i] < Nw ? colv[i] : (Nw - 1);
        wrow[i] = W + (size_t)colL * K + kboff;
    }

    v8f acc[4];
#pragma unroll
    for (int i = 0; i < 4; ++i)
#pragma unroll
        for (int j = 0; j < 8; ++j) acc[i][j] = 0.f;

    const unsigned short* arow = smemA + (size_t)(wm * 16 + lrow) * K + kaoff;

    // ---- software-pipelined K loop ----
    FragBF fa, fb[4];
    // prologue: load k0 = 0
    fa.u[0] = *reinterpret_cast<const uint4*>(arow);
    fa.u[1] = *reinterpret_cast<const uint4*>(arow + 16);
#pragma unroll
    for (int i = 0; i < 4; ++i) {
        fb[i].u[0] = *reinterpret_cast<const uint4*>(wrow[i]);
        fb[i].u[1] = *reinterpret_cast<const uint4*>(wrow[i] + 16);
    }

    for (int k0 = 0; k0 < K - 32; k0 += 32) {
        const int k1 = k0 + 32;
        FragBF fan, fbn[4];
        fan.u[0] = *reinterpret_cast<const uint4*>(arow + k1);
        fan.u[1] = *reinterpret_cast<const uint4*>(arow + k1 + 16);
#pragma unroll
        for (int i = 0; i < 4; ++i) {
            fbn[i].u[0] = *reinterpret_cast<const uint4*>(wrow[i] + k1);
            fbn[i].u[1] = *reinterpret_cast<const uint4*>(wrow[i] + k1 + 16);
        }

        acc[0] = __builtin_amdgcn_wmma_f32_16x16x32_bf16(false, fa.v, false, fb[0].v,
                                                         (short)0, acc[0], false, false);
        acc[1] = __builtin_amdgcn_wmma_f32_16x16x32_bf16(false, fa.v, false, fb[1].v,
                                                         (short)0, acc[1], false, false);
        acc[2] = __builtin_amdgcn_wmma_f32_16x16x32_bf16(false, fa.v, false, fb[2].v,
                                                         (short)0, acc[2], false, false);
        acc[3] = __builtin_amdgcn_wmma_f32_16x16x32_bf16(false, fa.v, false, fb[3].v,
                                                         (short)0, acc[3], false, false);

        fa = fan;
#pragma unroll
        for (int i = 0; i < 4; ++i) fb[i] = fbn[i];
    }

    // epilogue: last k-step
    acc[0] = __builtin_amdgcn_wmma_f32_16x16x32_bf16(false, fa.v, false, fb[0].v,
                                                     (short)0, acc[0], false, false);
    acc[1] = __builtin_amdgcn_wmma_f32_16x16x32_bf16(false, fa.v, false, fb[1].v,
                                                     (short)0, acc[1], false, false);
    acc[2] = __builtin_amdgcn_wmma_f32_16x16x32_bf16(false, fa.v, false, fb[2].v,
                                                     (short)0, acc[2], false, false);
    acc[3] = __builtin_amdgcn_wmma_f32_16x16x32_bf16(false, fa.v, false, fb[3].v,
                                                     (short)0, acc[3], false, false);

    // C layout: VGPR v -> M = m0 + (lane>=16 ? 8:0) + v ; N = nb + i*16 + (lane&15)
    const int rbase = m0 + (hi << 3);
#pragma unroll
    for (int i = 0; i < 4; ++i) {
        int col = colv[i];
        if (col >= N) continue;
        float bv = (col < Nreal) ? bias[col] : 0.f;
#pragma unroll
        for (int vv = 0; vv < 8; ++vv)
            C[(size_t)(rbase + vv) * N + col] = acc[i][vv] + bv;
    }
}

// ---------------- BN1 over lrelu(pre1) -> z0(=h0); BN2 of z0 half of si; init prev ----------------
__global__ __launch_bounds__(256)
void k_bn1(const float* __restrict__ pre, const float* __restrict__ g1, const float* __restrict__ b1,
           const float* __restrict__ g2, const float* __restrict__ b2,
           float* __restrict__ hA, unsigned short* __restrict__ h_bf,
           unsigned short* __restrict__ si_bf, int* __restrict__ prev) {
    const int c = blockIdx.x;        // 0..511
    const int r = threadIdx.x;       // 0..255
    float x = lrelu(pre[(size_t)r * HID + c]);
    float mu, va;
    stats256(x, mu, va);
    float z0 = g1[c] * (x - mu) * rsqrtf(va + EPSN) + b1[c];
    hA[(size_t)r * HID + c]  = z0;
    h_bf[(size_t)r * HID + c] = f32_to_bf16(z0);

    float mu2, va2;
    stats256(z0, mu2, va2);
    float siz = g2[HID + c] * (z0 - mu2) * rsqrtf(va2 + EPSN) + b2[HID + c];
    si_bf[(size_t)r * (2 * HID) + HID + c] = f32_to_bf16(siz);

    if (blockIdx.x == 0) prev[r] = SOS;
}

// ---------------- per-step: si[:, 0:H] = BN2(lrelu(emb[prev])) ----------------
__global__ __launch_bounds__(256)
void k_si(const float* __restrict__ emb, const int* __restrict__ prev,
          const float* __restrict__ g2, const float* __restrict__ b2,
          unsigned short* __restrict__ si_bf) {
    const int c = blockIdx.x;
    const int r = threadIdx.x;
    int p = prev[r];
    float x = lrelu(emb[(size_t)p * HID + c]);
    float mu, va;
    stats256(x, mu, va);
    float s = g2[c] * (x - mu) * rsqrtf(va + EPSN) + b2[c];
    si_bf[(size_t)r * (2 * HID) + c] = f32_to_bf16(s);
}

// ---------------- per-step: GRU gate combine + BN3(lrelu(h_new)) ----------------
__global__ __launch_bounds__(256)
void k_hup(const float* __restrict__ gi, const float* __restrict__ gh,
           const float* __restrict__ hA, const float* __restrict__ g3, const float* __restrict__ b3,
           float* __restrict__ hB, unsigned short* __restrict__ h_bf,
           unsigned short* __restrict__ o_bf) {
    const int c = blockIdx.x;        // 0..511
    const int r = threadIdx.x;       // 0..255
    const size_t gb = (size_t)r * G3;
    float ir = gi[gb + c],        iz = gi[gb + HID + c],  inn = gi[gb + 2 * HID + c];
    float hr = gh[gb + c],        hz = gh[gb + HID + c],  hn  = gh[gb + 2 * HID + c];
    float rg = sigmoidf_(ir + hr);
    float zg = sigmoidf_(iz + hz);
    float nn = tanhf(inn + rg * hn);
    float h  = hA[(size_t)r * HID + c];
    float hv = (1.f - zg) * nn + zg * h;
    hB[(size_t)r * HID + c]  = hv;
    h_bf[(size_t)r * HID + c] = f32_to_bf16(hv);

    float y = lrelu(hv);
    float mu, va;
    stats256(y, mu, va);
    float o = g3[c] * (y - mu) * rsqrtf(va + EPSN) + b3[c];
    o_bf[(size_t)r * HID + c] = f32_to_bf16(o);
}

// ---------------- per-step: gumbel + softmax + argmax ----------------
__global__ __launch_bounds__(256)
void k_softmax(float* __restrict__ logits,           // [B, VPAD] scratch (destroyed)
               const float* __restrict__ gu,          // [steps, B, VOCAB]
               const float* __restrict__ tptr,
               float* __restrict__ out,               // [B, steps, VOCAB]
               int* __restrict__ prev, int t, int steps) {
    const int r = blockIdx.x;
    const int tid = threadIdx.x;
    const float invT = 1.f / (*tptr);
    float* row = logits + (size_t)r * VPAD;
    const float* gur = gu + ((size_t)t * B + r) * VOCAB;

    float lmax = -3.4e38f;
    int   lidx = 0;
    for (int v = tid; v < VOCAB; v += 256) {
        float u = gur[v];
        float g = -logf(-logf(u + 1e-20f) + 1e-20f);
        float s = (row[v] + g) * invT;
        row[v] = s;
        if (s > lmax) { lmax = s; lidx = v; }
    }
    __shared__ float sm[256];
    __shared__ int   sx[256];
    sm[tid] = lmax; sx[tid] = lidx;
    __syncthreads();
#pragma unroll
    for (int off = 128; off > 0; off >>= 1) {
        if (tid < off) {
            if (sm[tid + off] > sm[tid] ||
                (sm[tid + off] == sm[tid] && sx[tid + off] < sx[tid])) {
                sm[tid] = sm[tid + off];
                sx[tid] = sx[tid + off];
            }
        }
        __syncthreads();
    }
    float M = sm[0];
    int amax = sx[0];
    __syncthreads();

    float lsum = 0.f;
    for (int v = tid; v < VOCAB; v += 256) {
        float e = expf(row[v] - M);
        row[v] = e;
        lsum += e;
    }
    float S = blockSum256(lsum);
    float inv = 1.f / S;
    float* orow = out + ((size_t)r * steps + t) * VOCAB;
    for (int v = tid; v < VOCAB; v += 256) orow[v] = row[v] * inv;
    if (tid == 0) prev[r] = amax;
}

// ---------------- host ----------------
extern "C" void kernel_launch(void* const* d_in, const int* in_sizes, int n_in,
                              void* d_out, int out_size, void* d_ws, size_t ws_size,
                              hipStream_t stream) {
    const float* z      = (const float*)d_in[0];
    const float* z2h_w  = (const float*)d_in[1];
    const float* z2h_b  = (const float*)d_in[2];
    const float* emb    = (const float*)d_in[3];
    const float* bn1_g  = (const float*)d_in[4];
    const float* bn1_b  = (const float*)d_in[5];
    const float* bn2_g  = (const float*)d_in[6];
    const float* bn2_b  = (const float*)d_in[7];
    const float* bn3_g  = (const float*)d_in[8];
    const float* bn3_b  = (const float*)d_in[9];
    const float* wih    = (const float*)d_in[10];
    const float* whh    = (const float*)d_in[11];
    const float* bih    = (const float*)d_in[12];
    const float* bhh    = (const float*)d_in[13];
    const float* h2o_w  = (const float*)d_in[14];
    const float* h2o_b  = (const float*)d_in[15];
    const float* gu     = (const float*)d_in[16];
    const float* tptr   = (const float*)d_in[18];
    float* out = (float*)d_out;

    int steps = in_sizes[16] / (B * VOCAB);
    if (steps <= 0) steps = 64;

    // workspace carve-up (256B aligned)
    uint8_t* base = (uint8_t*)d_ws;
    size_t off = 0;
    auto alloc = [&](size_t bytes) -> void* {
        void* p = base + off;
        off += (bytes + 255) & ~(size_t)255;
        return p;
    };
    unsigned short* z_bf    = (unsigned short*)alloc((size_t)B * NOISE * 2);
    unsigned short* z2h_bf  = (unsigned short*)alloc((size_t)HID * NOISE * 2);
    unsigned short* wih_bf  = (unsigned short*)alloc((size_t)G3 * (2 * HID) * 2);
    unsigned short* whh_bf  = (unsigned short*)alloc((size_t)G3 * HID * 2);
    unsigned short* h2o_bf  = (unsigned short*)alloc((size_t)VPAD * HID * 2);
    float*          pre1    = (float*)alloc((size_t)B * HID * 4);
    float*          hA      = (float*)alloc((size_t)B * HID * 4);
    float*          hB      = (float*)alloc((size_t)B * HID * 4);
    unsigned short* h_bf    = (unsigned short*)alloc((size_t)B * HID * 2);
    unsigned short* si_bf   = (unsigned short*)alloc((size_t)B * (2 * HID) * 2);
    unsigned short* o_bf    = (unsigned short*)alloc((size_t)B * HID * 2);
    float*          gi      = (float*)alloc((size_t)B * G3 * 4);
    float*          gh      = (float*)alloc((size_t)B * G3 * 4);
    float*          logits  = (float*)alloc((size_t)B * VPAD * 4);
    int*            prev    = (int*)alloc((size_t)B * 4);
    (void)ws_size;

    auto cvt = [&](const float* s, unsigned short* d, int n, int ntot) {
        k_cvt_bf16<<<(ntot + 255) / 256, 256, 0, stream>>>(s, d, n, ntot);
    };
    cvt(z,     z_bf,   B * NOISE,    B * NOISE);
    cvt(z2h_w, z2h_bf, HID * NOISE,  HID * NOISE);
    cvt(wih,   wih_bf, G3 * 2 * HID, G3 * 2 * HID);
    cvt(whh,   whh_bf, G3 * HID,     G3 * HID);
    cvt(h2o_w, h2o_bf, VOCAB * HID,  VPAD * HID);   // zero-pad rows 5000..5007

    auto gemm = [&](const unsigned short* Ax, const unsigned short* Wx, const float* bx,
                    float* Cx, int N, int Nw, int Nreal, int K) {
        dim3 g((N + 255) / 256, B / 32);
        size_t smem = (size_t)32 * K * 2;               // A panel in LDS
        k_gemm_bf16<<<g, 256, smem, stream>>>(Ax, Wx, bx, Cx, N, Nw, Nreal, K);
    };

    // pre1 = z @ z2h_w^T + z2h_b   (M=256, N=512, K=128)
    gemm(z_bf, z2h_bf, z2h_b, pre1, HID, HID, HID, NOISE);
    // z0 = BN1(lrelu(pre1)); h0 = z0; si[:,H:2H] = BN2(z0); prev = SOS
    k_bn1<<<HID, 256, 0, stream>>>(pre1, bn1_g, bn1_b, bn2_g, bn2_b, hA, h_bf, si_bf, prev);

    float* hCur = hA;
    float* hNxt = hB;
    for (int t = 0; t < steps; ++t) {
        // si[:,0:H] = BN2(lrelu(emb[prev]))
        k_si<<<HID, 256, 0, stream>>>(emb, prev, bn2_g, bn2_b, si_bf);
        // gi = si @ wih^T + bih   (M=256, N=1536, K=1024)
        gemm(si_bf, wih_bf, bih, gi, G3, G3, G3, 2 * HID);
        // gh = h @ whh^T + bhh    (M=256, N=1536, K=512)
        gemm(h_bf, whh_bf, bhh, gh, G3, G3, G3, HID);
        // GRU combine; h_new; o = BN3(lrelu(h_new))
        k_hup<<<HID, 256, 0, stream>>>(gi, gh, hCur, bn3_g, bn3_b, hNxt, h_bf, o_bf);
        // logits = o @ h2o_w^T + h2o_b  (M=256, N=5008(pad), K=512)
        gemm(o_bf, h2o_bf, h2o_b, logits, VPAD, VPAD, VOCAB, HID);
        // gumbel softmax -> out[:, t, :]; prev = argmax
        k_softmax<<<B, 256, 0, stream>>>(logits, gu, tptr, out, prev, t, steps);

        float* tmp = hCur; hCur = hNxt; hNxt = tmp;
    }
    (void)n_in; (void)out_size;
}
